// BinaryLinear_66151086293456
// MI455X (gfx1250) — 8-bit matrix operand order repaired and checked by reading, compiled, not run on hardware
//
#include <hip/hip_runtime.h>
#include <hip/hip_bf16.h>
#include <stdint.h>

typedef __attribute__((ext_vector_type(8))) int v8i;
typedef int v4i32 __attribute__((vector_size(16)));

#define M_DIM 8192
#define N_DIM 4096
#define K_DIM 4096
#define BM 128
#define BN 128
#define BK 64
#define LDS_STRIDE 80
#define TILE_BYTES (BM * LDS_STRIDE)
#define BUF_BYTES  (2 * TILE_BYTES)

#if defined(__HIP_DEVICE_COMPILE__) && \
    __has_builtin(__builtin_amdgcn_global_load_async_to_lds_b128)
#define ASYNC_COPY 1
#endif

#if defined(__HIP_DEVICE_COMPILE__)
template <int N>
__device__ __forceinline__ void wait_asynccnt() {
#if __has_builtin(__builtin_amdgcn_s_wait_asynccnt)
    __builtin_amdgcn_s_wait_asynccnt(N);
#else
    asm volatile("s_wait_asynccnt %0" ::"i"(N) : "memory");
#endif
}
#else
template <int N> __device__ __forceinline__ void wait_asynccnt() {}
#endif

#ifdef ASYNC_COPY
__device__ __forceinline__ void async_cp16(const signed char* g, signed char* l) {
    __builtin_amdgcn_global_load_async_to_lds_b128(
        (__attribute__((address_space(1))) v4i32*)(uintptr_t)g,
        (__attribute__((address_space(3))) v4i32*)(unsigned)(uintptr_t)l,
        0, 0);
}
#endif

__global__ void zero_sums_kernel(float* sums) {
    if (threadIdx.x < 2) sums[threadIdx.x] = 0.0f;
}

__global__ __launch_bounds__(256)
void bin_reduce_kernel(const float* __restrict__ in, signed char* __restrict__ out,
                       float* __restrict__ sum, long long n4) {
    long long i = (long long)blockIdx.x * blockDim.x + threadIdx.x;
    long long stride = (long long)gridDim.x * blockDim.x;
    const float4* in4 = (const float4*)in;
    char4* out4 = (char4*)out;
    float s = 0.0f;
    for (long long j = i; j < n4; j += stride) {
        float4 v = in4[j];
        s += fabsf(v.x) + fabsf(v.y) + fabsf(v.z) + fabsf(v.w);
        char4 c;
        c.x = (v.x >= 0.0f) ? (signed char)1 : (signed char)-1;
        c.y = (v.y >= 0.0f) ? (signed char)1 : (signed char)-1;
        c.z = (v.z >= 0.0f) ? (signed char)1 : (signed char)-1;
        c.w = (v.w >= 0.0f) ? (signed char)1 : (signed char)-1;
        out4[j] = c;
    }
    #pragma unroll
    for (int off = 16; off > 0; off >>= 1) s += __shfl_down(s, off, 32);
    __shared__ float red[8];
    int lane = threadIdx.x & 31;
    int wave = threadIdx.x >> 5;
    if (lane == 0) red[wave] = s;
    __syncthreads();
    if (wave == 0) {
        s = (lane < (int)(blockDim.x >> 5)) ? red[lane] : 0.0f;
        #pragma unroll
        for (int off = 4; off > 0; off >>= 1) s += __shfl_down(s, off, 32);
        if (lane == 0) atomicAdd(sum, s);
    }
}

__global__ __launch_bounds__(256)
void bgemm_wmma_kernel(const signed char* __restrict__ A,
                       const signed char* __restrict__ B,
                       const float* __restrict__ bias,
                       const float* __restrict__ sums,
                       float* __restrict__ out) {
    __shared__ signed char lds[2 * BUF_BYTES];

    const int tid  = threadIdx.x;
    const int lane = tid & 31;
    const int wave = tid >> 5;
    const int mBase = blockIdx.y * BM;
    const int nBase = blockIdx.x * BN;
    const int mWave = (wave >> 2) * 64;
    const int nWave = (wave & 3) * 32;

    const float scale = (sums[0] / (float)((long long)M_DIM * K_DIM)) *
                        (sums[1] / (float)((long long)N_DIM * K_DIM));

    v8i acc[4][2];
    const v8i vzero = {0, 0, 0, 0, 0, 0, 0, 0};
    #pragma unroll
    for (int mf = 0; mf < 4; mf++)
        #pragma unroll
        for (int nf = 0; nf < 2; nf++) acc[mf][nf] = vzero;

    const int lr = tid >> 2;
    const int lc = (tid & 3) * 16;
    const int sA0 = lr * LDS_STRIDE + lc;
    const int sA1 = (lr + 64) * LDS_STRIDE + lc;
    const int sB0 = TILE_BYTES + sA0;
    const int sB1 = TILE_BYTES + sA1;
    const signed char* gA0 = &A[(long long)(mBase + lr)      * K_DIM + lc];
    const signed char* gA1 = &A[(long long)(mBase + lr + 64) * K_DIM + lc];
    const signed char* gB0 = &B[(long long)(nBase + lr)      * K_DIM + lc];
    const signed char* gB1 = &B[(long long)(nBase + lr + 64) * K_DIM + lc];

    const int lrow  = lane & 15;
    const int halfA = (lane >> 4) * 8;
    const int halfB = (lane >> 4) * 8;
    int aBase[4], bBase[2];
    #pragma unroll
    for (int mf = 0; mf < 4; mf++)
        aBase[mf] = (mWave + mf * 16 + lrow) * LDS_STRIDE + halfA;
    #pragma unroll
    for (int nf = 0; nf < 2; nf++)
        bBase[nf] = TILE_BYTES + (nWave + nf * 16 + lrow) * LDS_STRIDE + halfB;

    const int NK = K_DIM / BK;

#ifdef ASYNC_COPY
#define STAGE(k0, dstOff)                                  \
    do {                                                   \
        async_cp16(gA0 + (k0), &lds[(dstOff) + sA0]);      \
        async_cp16(gA1 + (k0), &lds[(dstOff) + sA1]);      \
        async_cp16(gB0 + (k0), &lds[(dstOff) + sB0]);      \
        async_cp16(gB1 + (k0), &lds[(dstOff) + sB1]);      \
    } while (0)
#else
#define STAGE(k0, dstOff)                                              \
    do {                                                               \
        *(int4*)&lds[(dstOff) + sA0] = *(const int4*)(gA0 + (k0));     \
        *(int4*)&lds[(dstOff) + sA1] = *(const int4*)(gA1 + (k0));     \
        *(int4*)&lds[(dstOff) + sB0] = *(const int4*)(gB0 + (k0));     \
        *(int4*)&lds[(dstOff) + sB1] = *(const int4*)(gB1 + (k0));     \
    } while (0)
#endif

    STAGE(0, 0);

    int srcOff = 0;
    for (int kt = 0; kt < NK; ++kt) {
        const int kNext = ((kt + 1) & (NK - 1)) * BK;
        STAGE(kNext, srcOff ^ BUF_BYTES);

#ifdef ASYNC_COPY
        wait_asynccnt<4>();
#endif
        __syncthreads();

        v8i afrag[4], bfrag[2];
        #pragma unroll
        for (int mf = 0; mf < 4; mf++) {
            const signed char* p = &lds[srcOff + aBase[mf]];
            union { v8i v; int2 d[4]; } u;
            u.d[0] = *(const int2*)(p + 0);
            u.d[1] = *(const int2*)(p + 16);
            u.d[2] = *(const int2*)(p + 32);
            u.d[3] = *(const int2*)(p + 48);
            afrag[mf] = u.v;
        }
        #pragma unroll
        for (int nf = 0; nf < 2; nf++) {
            const signed char* p = &lds[srcOff + bBase[nf]];
            union { v8i v; int2 d[4]; } u;
            u.d[0] = *(const int2*)(p + 0);
            u.d[1] = *(const int2*)(p + 16);
            u.d[2] = *(const int2*)(p + 32);
            u.d[3] = *(const int2*)(p + 48);
            bfrag[nf] = u.v;
        }

        #pragma unroll
        for (int mf = 0; mf < 4; mf++)
            #pragma unroll
            for (int nf = 0; nf < 2; nf++)
                acc[mf][nf] = __builtin_amdgcn_wmma_i32_16x16x64_iu8(
                     true, afrag[mf],
                     true, bfrag[nf],
                    acc[mf][nf],  false,  false);

        __syncthreads();
        srcOff ^= BUF_BYTES;
    }
#undef STAGE

    #pragma unroll
    for (int mf = 0; mf < 4; mf++) {
        #pragma unroll
        for (int nf = 0; nf < 2; nf++) {
            const int col  = nBase + nWave + nf * 16 + lrow;
            const int row0 = mBase + mWave + mf * 16 + (lane >> 4) * 8;
            const float bv = bias[col];
            #pragma unroll
            for (int r = 0; r < 8; r++)
                out[(long long)(row0 + r) * N_DIM + col] =
                    scale * (float)acc[mf][nf][r] + bv;
        }
    }
}

extern "C" void kernel_launch(void* const* d_in, const int* in_sizes, int n_in,
                              void* d_out, int out_size, void* d_ws, size_t ws_size,
                              hipStream_t stream) {
    const float* x    = (const float*)d_in[0];
    const float* w    = (const float*)d_in[1];
    const float* bias = (const float*)d_in[2];
    float* out = (float*)d_out;

    float* sums = (float*)d_ws;
    signed char* Ab = (signed char*)d_ws + 256;
    signed char* Bb = Ab + (size_t)M_DIM * K_DIM;

    zero_sums_kernel<<<1, 32, 0, stream>>>(sums);
    bin_reduce_kernel<<<2048, 256, 0, stream>>>(x, Ab, &sums[0],
                                                (long long)M_DIM * K_DIM / 4);
    bin_reduce_kernel<<<1024, 256, 0, stream>>>(w, Bb, &sums[1],
                                                (long long)N_DIM * K_DIM / 4);

    dim3 grid(N_DIM / BN, M_DIM / BM);
    bgemm_wmma_kernel<<<grid, 256, 0, stream>>>(Ab, Bb, bias, sums, out);
}
